// PolicyNetClassifier_5574867550554
// MI455X (gfx1250) — matrix operand order repaired and checked by reading, not compiled, not run on hardware
//
#include <hip/hip_runtime.h>

// PolicyNetClassifier for MI455X (gfx1250, wave32, WMMA).
// Heavy GEMMs (output projection + recurrent LSTM matmuls) use
// V_WMMA_F32_16X16X32_BF16 with a hi/lo bf16 split (bf16x3 ~= f32 accuracy,
// f32 accumulate). Small xg pre-GEMMs use V_WMMA_F32_16X16X4_F32.
// Row max fused into the output GEMM (global_atomic_max_u32); two-phase
// radix-select top-k with candidate compaction; async global->LDS A-tile
// staging (global_load_async_to_lds_b128 + s_wait_asynccnt).

#define Bm 8
#define Sm 64
#define Km 20
#define Em 128
#define Hm 512
#define G4H 2048   // 4*H
#define CAP 16384  // top-k candidate capacity per row

typedef __attribute__((ext_vector_type(2)))  float  v2f;
typedef __attribute__((ext_vector_type(8)))  float  v8f;
typedef __attribute__((ext_vector_type(8)))  __bf16 v8bf;
typedef __attribute__((ext_vector_type(16))) __bf16 v16bf;
typedef int v4i __attribute__((vector_size(16)));   // matches async-LDS builtin
typedef __attribute__((address_space(1))) v4i* gptr_v4i;
typedef __attribute__((address_space(3))) v4i* lptr_v4i;

__device__ __forceinline__ v8f wmma4(v2f a, v2f b, v8f c) {
  return __builtin_amdgcn_wmma_f32_16x16x4_f32(false, a, false, b, (short)0, c,
                                               false, false);
}
__device__ __forceinline__ v8f wmma32(v16bf a, v16bf b, v8f c) {
  return __builtin_amdgcn_wmma_f32_16x16x32_bf16(false, a, false, b, (short)0, c,
                                                 false, false);
}
__device__ __forceinline__ v16bf cat8(v8bf a, v8bf b) {
  return __builtin_shufflevector(a, b, 0, 1, 2, 3, 4, 5, 6, 7,
                                 8, 9, 10, 11, 12, 13, 14, 15);
}
__device__ __forceinline__ v2f ld2(const float* p) {
  float2 t = *(const float2*)p;
  v2f v; v.x = t.x; v.y = t.y; return v;
}
__device__ __forceinline__ float sigm(float x) { return 1.0f / (1.0f + __expf(-x)); }

// -------------------------------------------------- hi/lo bf16 decomposition
__global__ void __launch_bounds__(256)
cvt_hilo(const float* __restrict__ W, __bf16* __restrict__ hi,
         __bf16* __restrict__ lo, long n) {
  long i = (long)blockIdx.x * 256 + threadIdx.x;
  if (i < n) {
    float f = W[i];
    __bf16 h = (__bf16)f;
    hi[i] = h;
    lo[i] = (__bf16)(f - (float)h);
  }
}

__global__ void init_rowmax(unsigned* __restrict__ rowmax) {
  rowmax[blockIdx.x * 256 + threadIdx.x] = 0u;
}

// ---------------------------------------------------------------- embedding
__global__ void embed_kernel(const int* __restrict__ inputs,
                             const int* __restrict__ mask,
                             const float* __restrict__ emb,
                             float* __restrict__ x) {
  int r = blockIdx.x;                       // r = b*S + s
  int id = inputs[r] * mask[r];
  x[(size_t)r * Em + threadIdx.x] = emb[(size_t)id * Em + threadIdx.x];
}

// ------------------------------------- C = A*W^T + b1 + b2 (f32 WMMA, small)
__global__ void __launch_bounds__(256)
gemm_bias_f32(const float* __restrict__ A, const float* __restrict__ W,
              const float* __restrict__ b1, const float* __restrict__ b2,
              float* __restrict__ C, int N, int K) {
  __shared__ float As[16 * 512];            // 32 KB, K <= 512
  const int tid = threadIdx.x;
  const int mbase = blockIdx.y * 16;
  for (int i = tid; i < 16 * K; i += 256) As[i] = A[(size_t)mbase * K + i];
  __syncthreads();

  const int wave = tid >> 5, lane = tid & 31;
  const int nt = blockIdx.x * 8 + wave;
  const int n15 = lane & 15;
  const int khalf = (lane >> 4) & 1;
  const int m = lane & 15;
  const int nglob = nt * 16 + n15;

  v8f acc{};
  const float bias = b1[nglob] + b2[nglob];
  for (int r = 0; r < 8; ++r) acc[r] = bias;

  const float* wrow = W + (size_t)nglob * K;
  for (int k0 = 0; k0 < K; k0 += 4) {
    const int kk = k0 + khalf * 2;
    acc = wmma4(ld2(&As[m * K + kk]), ld2(wrow + kk), acc);
  }
  for (int r = 0; r < 8; ++r) {
    int mm = r + khalf * 8;
    C[(size_t)(mbase + mm) * N + nglob] = acc[r];
  }
}

// ------------------------------------------------------------ sequential LSTM
__global__ void __launch_bounds__(1024)
lstm_seq(const float* __restrict__ xg,        // [512][2048], biases included
         const __bf16* __restrict__ Whi,      // [2048][512]
         const __bf16* __restrict__ Wlo,      // [2048][512]
         float* __restrict__ hout) {          // [B][S][H]
  __shared__ __align__(32) __bf16 Hhi[16 * Hm];   // rows 8..15 stay 0
  __shared__ __align__(32) __bf16 Hlo[16 * Hm];
  const int tid = threadIdx.x, wave = tid >> 5, lane = tid & 31;
  const int n15 = lane & 15;
  const int khalf = (lane >> 4) & 1;
  const int m = lane & 15;
  const int kb8 = khalf * 8;
  const int kb16 = khalf * 16;
  const int jcol = wave * 16 + n15;

  for (int i = tid; i < 16 * Hm; i += 1024) { Hhi[i] = (__bf16)0.0f; Hlo[i] = (__bf16)0.0f; }
  float c[8];
  for (int r = 0; r < 8; ++r) c[r] = 0.0f;

  const __bf16* wih = Whi + (size_t)jcol * Hm;
  const __bf16* wfh = Whi + (size_t)(Hm + jcol) * Hm;
  const __bf16* wgh = Whi + (size_t)(2 * Hm + jcol) * Hm;
  const __bf16* woh = Whi + (size_t)(3 * Hm + jcol) * Hm;
  const __bf16* wil = Wlo + (size_t)jcol * Hm;
  const __bf16* wfl = Wlo + (size_t)(Hm + jcol) * Hm;
  const __bf16* wgl = Wlo + (size_t)(2 * Hm + jcol) * Hm;
  const __bf16* wol = Wlo + (size_t)(3 * Hm + jcol) * Hm;
  __syncthreads();

  for (int s = 0; s < Sm; ++s) {
    v8f ai{}, af{}, ag{}, ao{};
    for (int r = 0; r < 8; ++r) {
      int b = r + khalf * 8;
      if (b < Bm) {
        size_t ro = ((size_t)b * Sm + s) * G4H;
        ai[r] = xg[ro + jcol];
        af[r] = xg[ro + Hm + jcol];
        ag[r] = xg[ro + 2 * Hm + jcol];
        ao[r] = xg[ro + 3 * Hm + jcol];
      }
    }
    for (int k0 = 0; k0 < Hm; k0 += 32) {
      const int ka = k0 + kb16;
      v16bf ah = cat8(*(const v8bf*)&Hhi[m * Hm + ka],
                      *(const v8bf*)&Hhi[m * Hm + ka + 8]);
      v16bf al = cat8(*(const v8bf*)&Hlo[m * Hm + ka],
                      *(const v8bf*)&Hlo[m * Hm + ka + 8]);
      const int kc = k0 + kb16;
      v16bf bh, bl;
      bh = *(const v16bf*)(wih + kc); bl = *(const v16bf*)(wil + kc);
      ai = wmma32(ah, bh, ai); ai = wmma32(ah, bl, ai); ai = wmma32(al, bh, ai);
      bh = *(const v16bf*)(wfh + kc); bl = *(const v16bf*)(wfl + kc);
      af = wmma32(ah, bh, af); af = wmma32(ah, bl, af); af = wmma32(al, bh, af);
      bh = *(const v16bf*)(wgh + kc); bl = *(const v16bf*)(wgl + kc);
      ag = wmma32(ah, bh, ag); ag = wmma32(ah, bl, ag); ag = wmma32(al, bh, ag);
      bh = *(const v16bf*)(woh + kc); bl = *(const v16bf*)(wol + kc);
      ao = wmma32(ah, bh, ao); ao = wmma32(ah, bl, ao); ao = wmma32(al, bh, ao);
    }
    __syncthreads();
    for (int r = 0; r < 8; ++r) {
      int b = r + khalf * 8;
      if (b < Bm) {
        float iv = sigm(ai[r]), fv = sigm(af[r]);
        float gv = tanhf(ag[r]), ov = sigm(ao[r]);
        c[r] = fv * c[r] + iv * gv;
        float hv = ov * tanhf(c[r]);
        __bf16 hh = (__bf16)hv;
        Hhi[b * Hm + jcol] = hh;
        Hlo[b * Hm + jcol] = (__bf16)(hv - (float)hh);
        hout[((size_t)b * Sm + s) * Hm + jcol] = hv;
      }
    }
    __syncthreads();
  }
}

// ---------------- logits = relu(h1 * Wout^T + bout), bf16x3 WMMA + fused row max
__global__ void __launch_bounds__(256)
out_gemm_relu(const float* __restrict__ Hin,    // [512][512]
              const __bf16* __restrict__ Whi,   // [V][512]
              const __bf16* __restrict__ Wlo,   // [V][512]
              const float* __restrict__ bout,   // [V]
              float* __restrict__ Out,          // [512][V]
              unsigned* __restrict__ rowmax,    // [512] f32 bits (relu >= 0)
              int V) {
  __shared__ union {
    float f[16 * Hm];                           // async staging (32 KB)
    struct { __bf16 hi[16 * Hm]; __bf16 lo[16 * Hm]; } b;
  } sm;
  __shared__ float rmax_s[8 * 16];
  const int tid = threadIdx.x;
  const int mbase = blockIdx.y * 16;

#if __has_builtin(__builtin_amdgcn_global_load_async_to_lds_b128) && \
    __has_builtin(__builtin_amdgcn_s_wait_asynccnt)
  // Stage the 16x512 f32 A-tile with async DMA, then convert in place.
  for (int c = tid; c < 16 * Hm / 4; c += 256) {
    __builtin_amdgcn_global_load_async_to_lds_b128(
        (gptr_v4i)(Hin + (size_t)mbase * Hm + c * 4),
        (lptr_v4i)&sm.f[c * 4], 0, 0);
  }
  __builtin_amdgcn_s_wait_asynccnt(0);
  __syncthreads();
  {
    float tmp[32];
    for (int t = 0; t < 32; ++t) tmp[t] = sm.f[t * 256 + tid];
    __syncthreads();
    for (int t = 0; t < 32; ++t) {
      int i = t * 256 + tid;
      __bf16 h = (__bf16)tmp[t];
      sm.b.hi[i] = h;
      sm.b.lo[i] = (__bf16)(tmp[t] - (float)h);
    }
  }
#else
  for (int i = tid; i < 16 * Hm; i += 256) {
    float f = Hin[(size_t)mbase * Hm + i];
    __bf16 h = (__bf16)f;
    sm.b.hi[i] = h;
    sm.b.lo[i] = (__bf16)(f - (float)h);
  }
#endif
  __syncthreads();

  const int wave = tid >> 5, lane = tid & 31;
  const long nt = (long)blockIdx.x * 8 + wave;
  const bool active = (nt * 16 < (long)V);   // wave-uniform
  const int n15 = lane & 15;
  const int khalf = (lane >> 4) & 1;
  const int m = lane & 15;
  float rvals[8];
  for (int r = 0; r < 8; ++r) rvals[r] = 0.0f;

  if (active) {
    const int kb8 = khalf * 8;
    const int kb16 = khalf * 16;
    long nglob = nt * 16 + n15;
    long ncl = nglob < V ? nglob : (long)V - 1;

    v8f acc{};
    const float bias = bout[ncl];
    for (int r = 0; r < 8; ++r) acc[r] = bias;

    const __bf16* wh = Whi + (size_t)ncl * Hm;
    const __bf16* wl = Wlo + (size_t)ncl * Hm;
    for (int k0 = 0; k0 < Hm; k0 += 32) {
      if ((k0 & 127) == 0) {
        __builtin_prefetch(wh + k0 + 128, 0, 1);
        __builtin_prefetch(wl + k0 + 128, 0, 1);
      }
      const int ka = k0 + kb16;
      v16bf ah = cat8(*(const v8bf*)&sm.b.hi[m * Hm + ka],
                      *(const v8bf*)&sm.b.hi[m * Hm + ka + 8]);
      v16bf al = cat8(*(const v8bf*)&sm.b.lo[m * Hm + ka],
                      *(const v8bf*)&sm.b.lo[m * Hm + ka + 8]);
      const int kc = k0 + kb16;
      v16bf bh = *(const v16bf*)(wh + kc);
      v16bf bl = *(const v16bf*)(wl + kc);
      acc = wmma32(ah, bh, acc);
      acc = wmma32(ah, bl, acc);
      acc = wmma32(al, bh, acc);
    }
    for (int r = 0; r < 8; ++r) {
      int mm = r + khalf * 8;
      float v = acc[r] > 0.0f ? acc[r] : 0.0f;
      rvals[r] = v;                          // clamped dup of col V-1 is harmless
      if (nglob < V) Out[(size_t)(mbase + mm) * V + nglob] = v;
    }
  }

  // per-row max: reduce 16 columns across the lane half, then across waves
  for (int r = 0; r < 8; ++r) {
    float v = rvals[r];
    for (int msk = 1; msk <= 8; msk <<= 1) v = fmaxf(v, __shfl_xor(v, msk, 32));
    if (n15 == 0) rmax_s[wave * 16 + r + khalf * 8] = v;
  }
  __syncthreads();
  if (tid < 16) {
    float mv = rmax_s[tid];
    for (int w = 1; w < 8; ++w) mv = fmaxf(mv, rmax_s[w * 16 + tid]);
    atomicMax(rowmax + mbase + tid, __float_as_uint(mv));
  }
}

// -------------------------------------------- per-row logsumexp (single pass)
__global__ void __launch_bounds__(256)
row_stats(const float* __restrict__ logits, const unsigned* __restrict__ rowmax,
          float* __restrict__ lse, int V) {
  const int r = blockIdx.x;
  const float* row = logits + (size_t)r * V;
  __shared__ float red[256];
  const int tid = threadIdx.x;
  const float mx = __uint_as_float(rowmax[r]);
  float sum = 0.0f;
  for (int i = tid; i < V; i += 256) sum += __expf(row[i] - mx);
  red[tid] = sum; __syncthreads();
  for (int o = 128; o > 0; o >>= 1) {
    if (tid < o) red[tid] += red[tid + o];
    __syncthreads();
  }
  if (tid == 0) lse[r] = mx + __logf(red[0]);
}

// -------------------------------------------------------------- loss (-logits)
__global__ void __launch_bounds__(256)
loss_kernel(const int* __restrict__ mask, const int* __restrict__ label,
            const float* __restrict__ ltype, const float* __restrict__ logits,
            const float* __restrict__ lse, float* __restrict__ out, int V) {
  const int b = blockIdx.x;
  __shared__ float red[256];
  const int tid = threadIdx.x;
  int vlen = 0;
  for (int s = 0; s < Sm; ++s) vlen += mask[b * Sm + s];
  float acc = 0.0f;
  for (int i = tid; i < Sm * Km; i += 256) {
    int s = i / Km, k = i % Km;
    float mf = (float)mask[b * Sm + s];
    float keep = (s >= vlen - 1) ? 1.0f : 0.0f;
    float lt = ltype[((size_t)b * Sm + s) * Km + k] * keep;
    int lab = label[((size_t)b * Sm + s) * Km + k];
    float lp = logits[(size_t)(b * Sm + s) * V + lab] - lse[b * Sm + s];
    acc += mf * lt * lp * (1.0f / (float)Km);
  }
  red[tid] = acc; __syncthreads();
  for (int o = 128; o > 0; o >>= 1) {
    if (tid < o) red[tid] += red[tid + o];
    __syncthreads();
  }
  if (tid == 0) out[b] = -red[0];
}

// ------------------- two-phase radix-select top-k (relu outputs >= 0)
__global__ void __launch_bounds__(256)
topk_kernel(const float* __restrict__ logits, const int* __restrict__ topk_ptr,
            float* __restrict__ out, int V,
            unsigned* __restrict__ cand_keys, int* __restrict__ cand_idx) {
  const int r = blockIdx.x;
  const float* row = logits + (size_t)r * V;
  const int tid = threadIdx.x;
  __shared__ unsigned hist[256];
  __shared__ int sh_kneed, sh_ctot, sh_ncand;
  __shared__ unsigned sh_prefix;
  __shared__ int cnt_gt, cnt_eq;
  const int Ktop = topk_ptr[0];
  unsigned* ck = cand_keys + (size_t)r * CAP;
  int* ci = cand_idx + (size_t)r * CAP;
  if (tid == 0) {
    sh_kneed = Ktop; sh_prefix = 0u; sh_ncand = 0; cnt_gt = 0; cnt_eq = 0;
  }
  hist[tid] = 0u;
  __syncthreads();

  // pass 1: full-row histogram of the top byte
  for (int i = tid; i < V; i += 256)
    atomicAdd(&hist[__float_as_uint(row[i]) >> 24], 1u);
  __syncthreads();
  if (tid == 0) {
    int need = sh_kneed;
    unsigned cumAbove = 0u;
    int sel = 0;
    for (int d = 255; d >= 0; --d) {
      unsigned h = hist[d];
      if (cumAbove + h >= (unsigned)need) { sel = d; break; }
      cumAbove += h;
    }
    sh_kneed = need - (int)cumAbove;
    sh_prefix = (unsigned)sel << 24;
    sh_ctot = (int)(cumAbove + hist[sel]);
  }
  __syncthreads();
  const unsigned sel1 = sh_prefix >> 24;
  const bool use_cand = (sh_ctot <= CAP);    // block-uniform

  if (use_cand) {
    // compact survivors (top byte >= sel1), then refine over candidates only
    for (int i = tid; i < V; i += 256) {
      unsigned key = __float_as_uint(row[i]);
      if ((key >> 24) >= sel1) {
        int p = atomicAdd(&sh_ncand, 1);
        if (p < CAP) { ck[p] = key; ci[p] = i; }
      }
    }
    __syncthreads();
    const int C = sh_ncand < CAP ? sh_ncand : CAP;
    for (int pass = 1; pass < 4; ++pass) {
      const int shift = 24 - 8 * pass;
      const unsigned hi_mask = 0xFFFFFFFFu << (shift + 8);
      hist[tid] = 0u;
      __syncthreads();
      const unsigned prefix = sh_prefix;
      for (int i = tid; i < C; i += 256) {
        unsigned key = ck[i];
        if ((key & hi_mask) == prefix) atomicAdd(&hist[(key >> shift) & 255u], 1u);
      }
      __syncthreads();
      if (tid == 0) {
        int need = sh_kneed;
        unsigned cumAbove = 0u;
        int sel = 0;
        for (int d = 255; d >= 0; --d) {
          unsigned h = hist[d];
          if (cumAbove + h >= (unsigned)need) { sel = d; break; }
          cumAbove += h;
        }
        sh_kneed = need - (int)cumAbove;
        sh_prefix = prefix | ((unsigned)sel << shift);
      }
      __syncthreads();
    }
    const unsigned T = sh_prefix;
    const int kEq = sh_kneed;
    float* orow = out + 8 + (size_t)r * Ktop;
    for (int i = tid; i < C; i += 256) {
      unsigned key = ck[i];
      if (key > T) {
        int p = atomicAdd(&cnt_gt, 1);
        if (p < Ktop) orow[p] = (float)ci[i];
      }
    }
    __syncthreads();
    int base = cnt_gt < Ktop ? cnt_gt : Ktop;
    for (int i = tid; i < C; i += 256) {
      unsigned key = ck[i];
      if (key == T) {
        int p = atomicAdd(&cnt_eq, 1);
        if (p < kEq && base + p < Ktop) orow[base + p] = (float)ci[i];
      }
    }
  } else {
    // fallback: refine over the full row
    for (int pass = 1; pass < 4; ++pass) {
      const int shift = 24 - 8 * pass;
      const unsigned hi_mask = 0xFFFFFFFFu << (shift + 8);
      hist[tid] = 0u;
      __syncthreads();
      const unsigned prefix = sh_prefix;
      for (int i = tid; i < V; i += 256) {
        unsigned key = __float_as_uint(row[i]);
        if ((key & hi_mask) == prefix) atomicAdd(&hist[(key >> shift) & 255u], 1u);
      }
      __syncthreads();
      if (tid == 0) {
        int need = sh_kneed;
        unsigned cumAbove = 0u;
        int sel = 0;
        for (int d = 255; d >= 0; --d) {
          unsigned h = hist[d];
          if (cumAbove + h >= (unsigned)need) { sel = d; break; }
          cumAbove += h;
        }
        sh_kneed = need - (int)cumAbove;
        sh_prefix = prefix | ((unsigned)sel << shift);
      }
      __syncthreads();
    }
    const unsigned T = sh_prefix;
    const int kEq = sh_kneed;
    float* orow = out + 8 + (size_t)r * Ktop;
    for (int i = tid; i < V; i += 256) {
      unsigned key = __float_as_uint(row[i]);
      if (key > T) {
        int p = atomicAdd(&cnt_gt, 1);
        if (p < Ktop) orow[p] = (float)i;
      }
    }
    __syncthreads();
    int base = cnt_gt < Ktop ? cnt_gt : Ktop;
    for (int i = tid; i < V; i += 256) {
      unsigned key = __float_as_uint(row[i]);
      if (key == T) {
        int p = atomicAdd(&cnt_eq, 1);
        if (p < kEq && base + p < Ktop) orow[base + p] = (float)i;
      }
    }
  }
}

// ------------------------------------------------------------------- launcher
extern "C" void kernel_launch(void* const* d_in, const int* in_sizes, int n_in,
                              void* d_out, int out_size, void* d_ws, size_t ws_size,
                              hipStream_t stream) {
  const int*   inputs = (const int*)d_in[0];
  const int*   mask   = (const int*)d_in[1];
  const int*   label  = (const int*)d_in[2];
  const float* ltype  = (const float*)d_in[3];
  const float* emb    = (const float*)d_in[4];
  const float* W_ih0  = (const float*)d_in[5];
  const float* W_hh0  = (const float*)d_in[6];
  const float* b_ih0  = (const float*)d_in[7];
  const float* b_hh0  = (const float*)d_in[8];
  const float* W_ih1  = (const float*)d_in[9];
  const float* W_hh1  = (const float*)d_in[10];
  const float* b_ih1  = (const float*)d_in[11];
  const float* b_hh1  = (const float*)d_in[12];
  const float* W_out  = (const float*)d_in[13];
  const float* b_out  = (const float*)d_in[14];
  const int*   topk   = (const int*)d_in[15];
  const int V = in_sizes[14];               // 127085
  float* out = (float*)d_out;

  // workspace layout (256B-aligned running cursor)
  char* ws = (char*)d_ws;
  size_t off = 0;
  auto take = [&](size_t bytes) {
    char* p = ws + off;
    off += (bytes + 255) & ~(size_t)255;
    return p;
  };
  float*    x      = (float*)take((size_t)Bm * Sm * Em * 4);
  float*    xg     = (float*)take((size_t)Bm * Sm * G4H * 4);
  float*    h0     = (float*)take((size_t)Bm * Sm * Hm * 4);
  float*    h1     = (float*)take((size_t)Bm * Sm * Hm * 4);
  float*    lse    = (float*)take((size_t)Bm * Sm * 4);
  unsigned* rowmax = (unsigned*)take((size_t)Bm * Sm * 4);
  __bf16*   whh0h  = (__bf16*)take((size_t)G4H * Hm * 2);
  __bf16*   whh0l  = (__bf16*)take((size_t)G4H * Hm * 2);
  __bf16*   whh1h  = (__bf16*)take((size_t)G4H * Hm * 2);
  __bf16*   whh1l  = (__bf16*)take((size_t)G4H * Hm * 2);
  __bf16*   wouth  = (__bf16*)take((size_t)V * Hm * 2);
  __bf16*   woutl  = (__bf16*)take((size_t)V * Hm * 2);
  unsigned* ckeys  = (unsigned*)take((size_t)Bm * Sm * CAP * 4);
  int*      cidx   = (int*)take((size_t)Bm * Sm * CAP * 4);
  float*    logits = (float*)take((size_t)Bm * Sm * V * 4);

  // one-time hi/lo decompositions
  {
    long n = (long)G4H * Hm;
    cvt_hilo<<<(unsigned)((n + 255) / 256), 256, 0, stream>>>(W_hh0, whh0h, whh0l, n);
    cvt_hilo<<<(unsigned)((n + 255) / 256), 256, 0, stream>>>(W_hh1, whh1h, whh1l, n);
    long nw = (long)V * Hm;
    cvt_hilo<<<(unsigned)((nw + 255) / 256), 256, 0, stream>>>(W_out, wouth, woutl, nw);
  }
  init_rowmax<<<(Bm * Sm) / 256, 256, 0, stream>>>(rowmax);

  embed_kernel<<<Bm * Sm, Em, 0, stream>>>(inputs, mask, emb, x);

  // layer 0
  gemm_bias_f32<<<dim3(G4H / 128, (Bm * Sm) / 16), 256, 0, stream>>>(
      x, W_ih0, b_ih0, b_hh0, xg, G4H, Em);
  lstm_seq<<<1, 1024, 0, stream>>>(xg, whh0h, whh0l, h0);

  // layer 1
  gemm_bias_f32<<<dim3(G4H / 128, (Bm * Sm) / 16), 256, 0, stream>>>(
      h0, W_ih1, b_ih1, b_hh1, xg, G4H, Hm);
  lstm_seq<<<1, 1024, 0, stream>>>(xg, whh1h, whh1l, h1);

  // output projection + relu + fused row max
  int ntiles = (V + 15) / 16;
  out_gemm_relu<<<dim3((ntiles + 7) / 8, (Bm * Sm) / 16), 256, 0, stream>>>(
      h1, wouth, woutl, b_out, logits, rowmax, V);

  row_stats<<<Bm * Sm, 256, 0, stream>>>(logits, rowmax, lse, V);
  loss_kernel<<<Bm, 256, 0, stream>>>(mask, label, ltype, logits, lse, out, V);
  topk_kernel<<<Bm * Sm, 256, 0, stream>>>(logits, topk, out, V, ckeys, cidx);
}
